// DeFormBlockv2_64321430224902
// MI455X (gfx1250) — matrix operand order repaired and checked by reading, not compiled, not run on hardware
//
#include <hip/hip_runtime.h>
#include <hip/hip_bf16.h>
#include <stdint.h>

typedef __attribute__((ext_vector_type(16))) __bf16 v16bf;
typedef __attribute__((ext_vector_type(8)))  float  v8f;

#define HH   128
#define WW   128
#define HW   16384
#define CIN  64
#define COUT 64
#define NB   4
#define KKT  9       // 3x3 taps
#define KDIM 576     // CIN * KKT
#define KT   18      // KDIM / 32  (bf16 WMMA K-steps)

__device__ __forceinline__ uint16_t f2bf(float f) {
  union { float f; uint32_t u; } v; v.f = f;
  uint32_t u = v.u;
  u += 0x7FFFu + ((u >> 16) & 1u);   // round-to-nearest-even
  return (uint16_t)(u >> 16);
}

// ---------------------------------------------------------------------------
// Pack w_def (Cout x Cin x 3 x 3 = 64 x 576) into bf16 WMMA A-fragments.
// Layout: apack[(mtile*KT + kk)*32 + lane][16], per ISA 16-bit A-matrix:
//   lanes 0-15  : M = lane,    K = kk*32 + {0..7, 16..23}
//   lanes 16-31 : M = lane-16, K = kk*32 + {8..15, 24..31}
// ---------------------------------------------------------------------------
__global__ void kPackA(const float* __restrict__ w_def, uint16_t* __restrict__ apack) {
  int t = blockIdx.x * blockDim.x + threadIdx.x;
  if (t >= 4 * KT * 32) return;
  int lane = t & 31;
  int kk   = (t >> 5) % KT;
  int m    = t / (KT * 32);
  int row  = m * 16 + (lane & 15);
  int base = kk * 32 + ((lane >> 4) ? 8 : 0);
  uint16_t* dst = apack + (size_t)t * 16;
#pragma unroll
  for (int j = 0; j < 8; ++j) dst[j]     = f2bf(w_def[(size_t)row * KDIM + base + j]);
#pragma unroll
  for (int j = 0; j < 8; ++j) dst[8 + j] = f2bf(w_def[(size_t)row * KDIM + base + 16 + j]);
}

// ---------------------------------------------------------------------------
// Offset conv: x (B,64,128,128) * w_off (27,64,3,3) -> om (B,27,128,128)
// One thread per output pixel; w_off staged through LDS in 8-channel chunks.
// ---------------------------------------------------------------------------
__global__ void kOffsetConv(const float* __restrict__ x, const float* __restrict__ w_off,
                            const float* __restrict__ b_off, float* __restrict__ om) {
  __shared__ float wsh[27 * 8 * 9];
  int tid = threadIdx.x;
  int pid = blockIdx.x * 128 + tid;
  int b   = pid >> 14;
  int rem = pid & (HW - 1);
  int y   = rem >> 7;
  int xq  = rem & 127;

  float acc[27];
#pragma unroll
  for (int o = 0; o < 27; ++o) acc[o] = b_off[o];

  for (int c0 = 0; c0 < CIN; c0 += 8) {
    __syncthreads();
    for (int i = tid; i < 27 * 8 * 9; i += 128) {
      int o = i / 72, r = i % 72, cc = r / 9, tt = r % 9;
      wsh[i] = w_off[((size_t)o * CIN + c0 + cc) * 9 + tt];
    }
    __syncthreads();
    for (int cc = 0; cc < 8; ++cc) {
      const float* xb = x + ((size_t)b * CIN + c0 + cc) * HW;
      float v[9];
#pragma unroll
      for (int tt = 0; tt < 9; ++tt) {
        int yy = y + tt / 3 - 1, xx = xq + tt % 3 - 1;
        bool ok = (yy >= 0) & (yy < HH) & (xx >= 0) & (xx < WW);
        v[tt] = ok ? xb[yy * WW + xx] : 0.0f;
      }
#pragma unroll
      for (int o = 0; o < 27; ++o) {
        float a = acc[o];
#pragma unroll
        for (int tt = 0; tt < 9; ++tt) a = fmaf(wsh[(o * 8 + cc) * 9 + tt], v[tt], a);
        acc[o] = a;
      }
    }
  }
  for (int o = 0; o < 27; ++o)
    om[((size_t)b * 27 + o) * HW + rem] = acc[o];
}

// ---------------------------------------------------------------------------
// Fused deformable sampling + WMMA GEMM.
// Workgroup: 16 pixels (one row segment) x 64 Cout; 4 waves, each wave owns a
// 16(Cout) x 16(pixel) tile -> 18x v_wmma_f32_16x16x32_bf16.
// ---------------------------------------------------------------------------
__global__ void __launch_bounds__(128) kDeform(
    const float* __restrict__ x, const float* __restrict__ om,
    const uint16_t* __restrict__ apack, const float* __restrict__ b_def,
    float* __restrict__ out) {
  __shared__ float mW[KKT][16][4];   // bilinear weights (mask folded in)
  __shared__ int   mI[KKT][16][4];   // clamped flat indices into one x-plane
  __shared__ __align__(32) uint16_t Bsh[KT * 32 * 16];  // 576x16 bf16 B tile

  int tid     = threadIdx.x;
  int tile    = blockIdx.x;            // 65536 tiles
  int b       = tile >> 10;            // 1024 tiles per batch image
  int pixbase = (tile & 1023) * 16;
  int y       = pixbase >> 7;
  int colbase = pixbase & 127;

  // ---- Phase A: per (tap, pixel) bilinear setup from om -------------------
  for (int e = tid; e < KKT * 16; e += 128) {
    int tap = e >> 4, p = e & 15;
    int pix = pixbase + p;
    float oy = om[((size_t)b * 27 + tap) * HW + pix];
    float ox = om[((size_t)b * 27 + 9 + tap) * HW + pix];
    float mk = om[((size_t)b * 27 + 18 + tap) * HW + pix];
    mk = 1.0f / (1.0f + __expf(-mk));           // sigmoid mask
    int kh = tap / 3, kw = tap % 3;
    float sy = oy + (float)(y - 1 + kh);
    float sx = ox + (float)(colbase + p - 1 + kw);
    float y0f = floorf(sy), x0f = floorf(sx);
    int y0 = (int)y0f, x0 = (int)x0f;
    float wy = sy - y0f, wx = sx - x0f;
    float ww[4] = { (1.0f - wy) * (1.0f - wx) * mk, (1.0f - wy) * wx * mk,
                    wy * (1.0f - wx) * mk,          wy * wx * mk };
#pragma unroll
    for (int q = 0; q < 4; ++q) {
      int yi = y0 + (q >> 1), xi = x0 + (q & 1);
      bool ok = (yi >= 0) & (yi < HH) & (xi >= 0) & (xi < WW);
      int yc = min(max(yi, 0), HH - 1), xc = min(max(xi, 0), WW - 1);
      mI[tap][p][q] = yc * WW + xc;
      mW[tap][p][q] = ok ? ww[q] : 0.0f;
    }
  }
  __syncthreads();

  // ---- Phase B: gather + interpolate -> bf16 B tile in WMMA B layout ------
  // K index = c*9 + tap (matches w_def.reshape(Cout, Cin*KK)); B fragment:
  // lanes 0-15 -> cols, K = kk*32+{0..7,16..23}; lanes 16-31 -> K = kk*32+{8..15,24..31}.
  for (int e = tid; e < CIN * KKT * 16; e += 128) {   // 9216 / 128 = 72 exact
    int c = e / 144, r = e % 144, tap = r >> 4, p = r & 15;
    const float* xb = x + ((size_t)b * CIN + c) * HW;
    const float* w  = mW[tap][p];
    const int*   ii = mI[tap][p];
    float v = w[0] * xb[ii[0]] + w[1] * xb[ii[1]] + w[2] * xb[ii[2]] + w[3] * xb[ii[3]];
    int K = c * 9 + tap;
    int kk = K >> 5, kpos = K & 31;
    int lane = ((kpos >> 3) & 1) ? (p + 16) : p;
    Bsh[(kk * 32 + lane) * 16 + ((kpos & 7) | ((kpos >> 4) << 3))] = f2bf(v);
  }
  __syncthreads();

  // ---- Phase C: 18-step WMMA K-loop, one 16x16 tile per wave --------------
  int wave = tid >> 5, lane = tid & 31;
  v8f acc = {};
  const uint16_t* abase = apack + ((size_t)wave * KT * 32 + lane) * 16;
#pragma unroll 2
  for (int kk = 0; kk < KT; ++kk) {
    v16bf a  = *reinterpret_cast<const v16bf*>(abase + (size_t)kk * 32 * 16);
    v16bf bm = *reinterpret_cast<const v16bf*>(&Bsh[(kk * 32 + lane) * 16]);
    acc = __builtin_amdgcn_wmma_f32_16x16x32_bf16(false, a, false, bm,
                                                  (short)0, acc, false, false);
  }
  // C/D layout: VGPR r -> M = r (lanes 0-15) or r+8 (lanes 16-31); N = lane%16
  int moff = (lane >> 4) * 8;
  int col  = lane & 15;
#pragma unroll
  for (int r = 0; r < 8; ++r) {
    int mrow = wave * 16 + moff + r;
    out[((size_t)b * COUT + mrow) * HW + pixbase + col] = acc[r] + b_def[mrow];
  }
}

// ---------------------------------------------------------------------------
// BN statistics: deterministic two-level reduction (no float atomics).
// Grid = 64 channels x 16 segments; partials -> ws, reduced in kFinalize.
// ---------------------------------------------------------------------------
__global__ void kStats(const float* __restrict__ out,
                       float* __restrict__ part1, float* __restrict__ part2) {
  __shared__ float s1[256], s2[256];
  int c = blockIdx.x >> 4, seg = blockIdx.x & 15;
  float a = 0.0f, a2 = 0.0f;
  for (int j = seg * 4096 + threadIdx.x; j < (seg + 1) * 4096; j += 256) {
    int b = j >> 14, i = j & (HW - 1);   // per-channel linear index over (B, HW)
    float v = out[((size_t)b * COUT + c) * HW + i];
    a += v; a2 += v * v;
  }
  s1[threadIdx.x] = a; s2[threadIdx.x] = a2;
  __syncthreads();
  for (int st = 128; st > 0; st >>= 1) {
    if (threadIdx.x < st) {
      s1[threadIdx.x] += s1[threadIdx.x + st];
      s2[threadIdx.x] += s2[threadIdx.x + st];
    }
    __syncthreads();
  }
  if (threadIdx.x == 0) { part1[blockIdx.x] = s1[0]; part2[blockIdx.x] = s2[0]; }
}

__global__ void kFinalize(const float* __restrict__ part1, const float* __restrict__ part2,
                          const float* __restrict__ gamma, const float* __restrict__ beta,
                          float* __restrict__ scale, float* __restrict__ shift) {
  int c = threadIdx.x;
  if (c >= COUT) return;
  float s = 0.0f, s2 = 0.0f;
  for (int seg = 0; seg < 16; ++seg) { s += part1[c * 16 + seg]; s2 += part2[c * 16 + seg]; }
  const float invN = 1.0f / (float)(NB * HW);
  float mean = s * invN;
  float var  = s2 * invN - mean * mean;
  float sc   = gamma[c] * rsqrtf(var + 1e-5f);
  scale[c] = sc;
  shift[c] = beta[c] - mean * sc;
}

__global__ void kBNRelu(float* __restrict__ out, const float* __restrict__ scale,
                        const float* __restrict__ shift) {
  size_t i = (size_t)blockIdx.x * 256 + threadIdx.x;
  int c = (int)((i >> 14) & 63);
  float v = out[i] * scale[c] + shift[c];
  out[i] = fmaxf(v, 0.0f);
}

// ---------------------------------------------------------------------------
extern "C" void kernel_launch(void* const* d_in, const int* in_sizes, int n_in,
                              void* d_out, int out_size, void* d_ws, size_t ws_size,
                              hipStream_t stream) {
  const float* x     = (const float*)d_in[0];
  const float* w_off = (const float*)d_in[1];
  const float* b_off = (const float*)d_in[2];
  const float* w_def = (const float*)d_in[3];
  const float* b_def = (const float*)d_in[4];
  const float* gamma = (const float*)d_in[5];
  const float* beta  = (const float*)d_in[6];
  float* out = (float*)d_out;

  // Workspace layout (~7.1 MB total):
  float*    om    = (float*)d_ws;                       // 27*4*16384 floats
  float*    part1 = om + (size_t)27 * NB * HW;          // 1024
  float*    part2 = part1 + 1024;                       // 1024
  float*    scale = part2 + 1024;                       // 64
  float*    shift = scale + 64;                         // 64
  uint16_t* apack = (uint16_t*)(shift + 64 + 32);       // 4*18*32*16 bf16 (32B aligned)

  kPackA     <<<9,     256, 0, stream>>>(w_def, apack);
  kOffsetConv<<<512,   128, 0, stream>>>(x, w_off, b_off, om);
  kDeform    <<<65536, 128, 0, stream>>>(x, om, apack, b_def, out);
  kStats     <<<1024,  256, 0, stream>>>(out, part1, part2);
  kFinalize  <<<1,      64, 0, stream>>>(part1, part2, gamma, beta, scale, shift);
  kBNRelu    <<<16384, 256, 0, stream>>>(out, scale, shift);
}
